// GotenNet_20907900797507
// MI455X (gfx1250) — matrix operand order repaired and checked by reading, not compiled, not run on hardware
//
#include <hip/hip_runtime.h>

#define B_   2
#define N_   128
#define D_   128
#define DE_  64
#define H_   8
#define DH_  32
#define DI_  256
#define DMI_ 512
#define S_   5
#define SDI_ 1280
#define FFD_ 512
#define BN_  (B_*N_)
#define BNN_ (B_*N_*N_)

typedef float v2f __attribute__((ext_vector_type(2)));
typedef float v8f __attribute__((ext_vector_type(8)));
typedef __bf16 bf16;
typedef bf16 v8bf  __attribute__((ext_vector_type(8)));
typedef bf16 v16bf __attribute__((ext_vector_type(16)));

__device__ __forceinline__ v8f wmma4(v2f a, v2f b, v8f c) {
  // D = A(16x4 f32) * B(4x16 f32) + C  (full-precision path, small GEMMs)
  return __builtin_amdgcn_wmma_f32_16x16x4_f32(false, a, false, b, (short)0, c, false, false);
}
__device__ __forceinline__ v8f wmma_bf(v16bf a, v16bf b, v8f c) {
  // D = A(16x32 bf16) * B(32x16 bf16) + C(f32)  (hot path, 8x FLOP/instr)
  return __builtin_amdgcn_wmma_f32_16x16x32_bf16(false, a, false, b, (short)0, c, false, false);
}
__device__ __forceinline__ v16bf cat16(v8bf lo, v8bf hi) {
  return __builtin_shufflevector(lo, hi, 0,1,2,3,4,5,6,7,8,9,10,11,12,13,14,15);
}
// A-fragment (16-bit A 16x32): lane=half*16+lm reads row `arow`,
// K = k0 + 16*half + 0..15 (contiguous, matches B fragments)
__device__ __forceinline__ v16bf load_a_bf(const bf16* row, int k0, int half) {
  return *(const v16bf*)(row + k0 + 16 * half);
}
__device__ __forceinline__ float silu_f(float x) { return x / (1.f + __expf(-x)); }

// ---------------------------------------------------------------------------
// Fused: ek tile = silu( t_bf(16j x 128) @ WekT_bf ) ; sim_den += q * sum_j ek
// grid = (N/16, SDI/16, B*N), 1 wave/block.  bf16 WMMA, K-step 32.
// ---------------------------------------------------------------------------
__global__ __launch_bounds__(32)
void simden_kernel(const bf16* __restrict__ tb, const bf16* __restrict__ WekT,
                   const float* __restrict__ q, float* __restrict__ sden)
{
  const int lane = threadIdx.x & 31, half = lane >> 4, lm = lane & 15;
  const int jt = blockIdx.x * 16;
  const int c0 = blockIdx.y * 16;
  const int bi = blockIdx.z, b = bi >> 7, i = bi & 127;
  const bf16* Ab = tb + (long)(b * N_ + i) * N_ * D_ + (long)(jt + lm) * D_;
  const bf16* Bb = WekT + (long)(c0 + lm) * D_;   // WekT[n][k], contiguous in k
  __builtin_prefetch(Ab, 0, 3);

  v8f acc = {};
#pragma unroll
  for (int k0 = 0; k0 < D_; k0 += 32) {
    v16bf a  = load_a_bf(Ab, k0, half);
    v16bf bv = *(const v16bf*)(Bb + k0 + 16 * half);   // K = k0+16*half+0..15
    acc = wmma_bf(a, bv, acc);
  }
  const int s = c0 >> 8, rem = c0 & 255, hh = rem >> 5, d0 = rem & 31;
  float qv = q[(long)(b * N_ + i) * DI_ + hh * DH_ + d0 + lm];
  float ssum = 0.f;
#pragma unroll
  for (int r = 0; r < 8; ++r) ssum += silu_f(acc[r]);
  float p = qv * ssum;
  for (int off = 16; off > 0; off >>= 1) p += __shfl_down(p, off, 32);
  if (lane == 0)
    atomicAdd(&sden[((long)(b * H_ + hh) * N_ + i) * S_ + s], p);
}

// ---------------------------------------------------------------------------
// Fused pass 2 (bf16 WMMA hot path): per (b,i, 16-wide j tile), for each s:
//   ev tile (16j x 256k) = t_bf @ WevT_bf           -> LDS (f32)
//   A[j,k]  = attn*v + ev*pv                         -> LDS (bf16)
//   out(16j x 128e) = A @ WcombT_bf                  (bf16 WMMA, f32 acc)
//   reduce over j into h2 / x1a / x2a via atomics
// The t-tile A-fragments are loop-invariant across (s, cti): load ONCE.
// block = 256 threads = 8 waves;  grid = (N/16, B*N)
// ---------------------------------------------------------------------------
__global__ __launch_bounds__(256)
void pass2_kernel(const bf16* __restrict__ tb, const bf16* __restrict__ WevT,
                  const bf16* __restrict__ WcombT,
                  const float* __restrict__ snum, const float* __restrict__ sden,
                  const float* __restrict__ v, const float* __restrict__ pv,
                  const float* __restrict__ x1, const float* __restrict__ x2,
                  const float* __restrict__ r1, const float* __restrict__ r2,
                  float* __restrict__ h2, float* __restrict__ x1a, float* __restrict__ x2a)
{
  __shared__ float evs[16 * 256];    // ev tile, f32
  __shared__ bf16  Asb[16 * 256];    // A matrix for out-GEMM, bf16
  const int tid = threadIdx.x, w = tid >> 5, lane = tid & 31, half = lane >> 4, lm = lane & 15;
  const int jt = blockIdx.x * 16;
  const int bi = blockIdx.y, b = bi >> 7, i = bi & 127;
  const bf16* trow = tb + (long)(b * N_ + i) * N_ * D_;
  const long rowi = (long)(b * N_ + i);

  // hoisted t-tile A fragments (reused by all 10 ev-GEMMs of this block)
  const bf16* Ar = trow + (long)(jt + lm) * D_;
  v16bf afrag[4];
#pragma unroll
  for (int kk = 0; kk < 4; ++kk) afrag[kk] = load_a_bf(Ar, kk * 32, half);

  for (int s = 0; s < S_; ++s) {
    // --- ev tile for this s (16 column sub-tiles; wave w does 2) ---
#pragma unroll
    for (int cti = 0; cti < 2; ++cti) {
      const int lc0 = (w * 2 + cti) * 16;            // local col in [0,256)
      const int c0  = s * DI_ + lc0;                 // global col in Wev
      const bf16* Br = WevT + (long)(c0 + lm) * D_;
      v8f acc = {};
#pragma unroll
      for (int k0 = 0; k0 < 4; ++k0) {
        v16bf bv = *(const v16bf*)(Br + k0 * 32 + 16 * half);
        acc = wmma_bf(afrag[k0], bv, acc);
      }
#pragma unroll
      for (int r = 0; r < 8; ++r) evs[(r + 8 * half) * 256 + lc0 + lm] = acc[r];
    }
    __syncthreads();

    // --- build A = attn*v + ev*pv  (bf16 for the matrix pipe) ---
    for (int e = tid; e < 16 * 256; e += 256) {
      const int j = e >> 8, kk = e & 255, hh = kk >> 5;
      const long bhi = (long)(b * H_ + hh) * N_ + i;
      float attn = snum[bhi * N_ + jt + j] / sden[bhi * S_ + s];
      const long vidx = (long)(b * N_ + jt + j) * SDI_ + s * DI_ + kk;
      Asb[e] = (bf16)(attn * v[vidx] + evs[e] * pv[vidx]);
    }
    __syncthreads();

    // --- out tile: A(16x256) @ Wcomb(256x128); wave w owns cols e0..e0+15 ---
    const int e0 = w * 16;
    v8f acc = {};
#pragma unroll
    for (int k0 = 0; k0 < DI_; k0 += 32) {
      v16bf a  = load_a_bf(&Asb[lm * 256], k0, half);
      v16bf bv = *(const v16bf*)(WcombT + (long)(e0 + lm) * DI_ + k0 + 16 * half);
      acc = wmma_bf(a, bv, acc);
    }

    // --- reductions over j (rows) per output column e ---
    const int e = e0 + lm;
    if (s == 0) {
      float p = 0.f;
#pragma unroll
      for (int r = 0; r < 8; ++r) p += acc[r];
      p += __shfl_down(p, 16, 32);
      if (lane < 16) atomicAdd(&h2[rowi * D_ + e], p);
    } else if (s == 1 || s == 3) {
      float pm[3] = {0.f, 0.f, 0.f};
#pragma unroll
      for (int r = 0; r < 8; ++r) {
        const int j = jt + r + 8 * half;
        const float c = acc[r];
        if (s == 1) {
          const float* rr = &r1[(rowi * N_ + j) * 3];
#pragma unroll
          for (int m = 0; m < 3; ++m) pm[m] += rr[m] * c;
        } else {
          const float* xx = &x1[((long)(b * N_ + j) * D_ + e) * 3];
#pragma unroll
          for (int m = 0; m < 3; ++m) pm[m] += xx[m] * c;
        }
      }
#pragma unroll
      for (int m = 0; m < 3; ++m) pm[m] += __shfl_down(pm[m], 16, 32);
      if (lane < 16)
#pragma unroll
        for (int m = 0; m < 3; ++m) atomicAdd(&x1a[(rowi * D_ + e) * 3 + m], pm[m]);
    } else { // s==2 || s==4
      float pm[5] = {0.f, 0.f, 0.f, 0.f, 0.f};
#pragma unroll
      for (int r = 0; r < 8; ++r) {
        const int j = jt + r + 8 * half;
        const float c = acc[r];
        if (s == 2) {
          const float* rr = &r2[(rowi * N_ + j) * 5];
#pragma unroll
          for (int m = 0; m < 5; ++m) pm[m] += rr[m] * c;
        } else {
          const float* xx = &x2[((long)(b * N_ + j) * D_ + e) * 5];
#pragma unroll
          for (int m = 0; m < 5; ++m) pm[m] += xx[m] * c;
        }
      }
#pragma unroll
      for (int m = 0; m < 5; ++m) pm[m] += __shfl_down(pm[m], 16, 32);
      if (lane < 16)
#pragma unroll
        for (int m = 0; m < 5; ++m) atomicAdd(&x2a[(rowi * D_ + e) * 5 + m], pm[m]);
    }
    __syncthreads();
  }
}

// ---------------------------------------------------------------------------
// Generic strided + batched fp32 WMMA GEMM (one wave == one 16x16 C tile).
// flags: bit0 = silu epilogue, bit1 = accumulate Cin (same layout as C)
// ---------------------------------------------------------------------------
__global__ __launch_bounds__(32)
void wmma_gemm_f32(const float* __restrict__ A, const float* __restrict__ Bm,
                   const float* __restrict__ bias, const float* __restrict__ Cin,
                   float* __restrict__ C,
                   int M, int Nn, int K,
                   int aRS, int aCS, int bRS, int bCS, int cRS, int cCS,
                   long aB0, long aB1, long bB0, long bB1, long cB0, long cB1,
                   int zdiv, int flags)
{
  const int lane = threadIdx.x & 31;
  const int half = lane >> 4;
  const int lm   = lane & 15;
  const int mi = blockIdx.x * 16;
  const int nj = blockIdx.y * 16;
  const int z  = blockIdx.z;
  const int z0 = z / zdiv, z1 = z % zdiv;
  const float* Ab   = A  + z0 * aB0 + z1 * aB1;
  const float* Bb   = Bm + z0 * bB0 + z1 * bB1;
  float*       Cb   = C  + z0 * cB0 + z1 * cB1;
  const float* Cinb = Cin ? (Cin + z0 * cB0 + z1 * cB1) : (const float*)0;

  int arow = mi + lm; float am = (arow < M)  ? 1.f : 0.f; if (arow >= M)  arow = M - 1;
  int bcol = nj + lm; float bm = (bcol < Nn) ? 1.f : 0.f; if (bcol >= Nn) bcol = Nn - 1;

  v8f acc = {};
  for (int k0 = 0; k0 < K; k0 += 4) {
    const int ka = k0 + 2 * half;
    v2f a, b;
    a.x = Ab[(long)arow * aRS + (long)(ka + 0) * aCS] * am;
    a.y = Ab[(long)arow * aRS + (long)(ka + 1) * aCS] * am;
    b.x = Bb[(long)(ka + 0) * bRS + (long)bcol * bCS] * bm;
    b.y = Bb[(long)(ka + 1) * bRS + (long)bcol * bCS] * bm;
    acc = wmma4(a, b, acc);
  }

  if (nj + lm >= Nn) return;
  const int col = nj + lm;
#pragma unroll
  for (int r = 0; r < 8; ++r) {
    int row = mi + r + 8 * half;
    if (row >= M) continue;
    float val = acc[r];
    if (flags & 2) val += Cinb[(long)row * cRS + (long)col * cCS];
    if (bias) val += bias[col];
    if (flags & 1) val = silu_f(val);
    Cb[(long)row * cRS + (long)col * cCS] = val;
  }
}

// ---------------------------------------------------------------------------
// LayerNorm: hi = LN(h)*g_i, hj = LN(h)*g_j
// ---------------------------------------------------------------------------
__global__ void ln_kernel(const float* __restrict__ h, const float* __restrict__ gi,
                          const float* __restrict__ gj,
                          float* __restrict__ hi, float* __restrict__ hj)
{
  __shared__ float red[D_];
  const int row = blockIdx.x, d = threadIdx.x;
  float x = h[row * D_ + d];
  red[d] = x; __syncthreads();
  for (int o = D_/2; o > 0; o >>= 1) { if (d < o) red[d] += red[d + o]; __syncthreads(); }
  float mu = red[0] / (float)D_; __syncthreads();
  float c = x - mu;
  red[d] = c * c; __syncthreads();
  for (int o = D_/2; o > 0; o >>= 1) { if (d < o) red[d] += red[d + o]; __syncthreads(); }
  float inv = rsqrtf(red[0] / (float)D_ + 1e-5f);
  hi[row * D_ + d] = c * inv * gi[d];
  hj[row * D_ + d] = c * inv * gj[d];
}

// ---------------------------------------------------------------------------
// w_ij[b,i,j,e]
// ---------------------------------------------------------------------------
__global__ void wij_kernel(const float* __restrict__ q1, const float* __restrict__ k1,
                           const float* __restrict__ q2, const float* __restrict__ k2,
                           float* __restrict__ wij)
{
  int idx = blockIdx.x * 256 + threadIdx.x;        // B*N*N*128 total
  int e = idx & 127, j = (idx >> 7) & 127, i = (idx >> 14) & 127, b = idx >> 21;
  float s = 0.f;
  if (e < DE_) {
    const float* qa = &q1[((long)(b * N_ + i) * DE_ + e) * 3];
    const float* ka = &k1[((long)(b * N_ + j) * DE_ + e) * 3];
#pragma unroll
    for (int m = 0; m < 3; ++m) s += qa[m] * ka[m];
  } else {
    int ee = e - DE_;
    const float* qa = &q2[((long)(b * N_ + i) * DE_ + ee) * 5];
    const float* ka = &k2[((long)(b * N_ + j) * DE_ + ee) * 5];
#pragma unroll
    for (int m = 0; m < 5; ++m) s += qa[m] * ka[m];
  }
  wij[idx] = s;
}

// ---------------------------------------------------------------------------
// init: h2 = h ; x1a = x2a = sim_den = 0
// ---------------------------------------------------------------------------
__global__ void init_kernel(const float* __restrict__ h, float* __restrict__ h2,
                            float* __restrict__ x1a, float* __restrict__ x2a,
                            float* __restrict__ sden)
{
  int idx = blockIdx.x * 256 + threadIdx.x;
  if (idx < BN_ * D_)          h2[idx]  = h[idx];
  if (idx < BN_ * D_ * 3)      x1a[idx] = 0.f;
  if (idx < BN_ * D_ * 5)      x2a[idx] = 0.f;
  if (idx < B_ * H_ * N_ * S_) sden[idx] = 0.f;
}

// ---------------------------------------------------------------------------
// f32 -> bf16 elementwise (for t)
// ---------------------------------------------------------------------------
__global__ void cvt_bf16_kernel(const float* __restrict__ src, bf16* __restrict__ dst, long n)
{
  long idx = (long)blockIdx.x * 256 + threadIdx.x;
  if (idx < n) dst[idx] = (bf16)src[idx];
}

// ---------------------------------------------------------------------------
// transpose + convert weight: dst[c*K + k] = (bf16) src[k*Nc + c]
// ---------------------------------------------------------------------------
__global__ void convT_kernel(const float* __restrict__ src, bf16* __restrict__ dst,
                             int K, int Nc)
{
  int idx = blockIdx.x * 256 + threadIdx.x;
  if (idx >= K * Nc) return;
  int k = idx / Nc, c = idx % Nc;
  dst[(long)c * K + k] = (bf16)src[idx];
}

// ---------------------------------------------------------------------------
__global__ void invnorm_kernel(const float* __restrict__ proj, float* __restrict__ inv, int mc)
{
  int idx = blockIdx.x * 256 + threadIdx.x;      // B*N*D
  if (idx >= BN_ * D_) return;
  float s = 1e-12f;
  for (int m = 0; m < mc; ++m) { float p = proj[(long)idx * mc + m]; s += p * p; }
  inv[idx] = sqrtf(s);
}

// ---------------------------------------------------------------------------
__global__ void final_kernel(const float* __restrict__ h2,
                             const float* __restrict__ moa, const float* __restrict__ mob,
                             const float* __restrict__ x1a, const float* __restrict__ x2a,
                             const float* __restrict__ p1, const float* __restrict__ p2,
                             float* __restrict__ out)
{
  int idx = blockIdx.x * 256 + threadIdx.x;      // B*N*D
  if (idx >= BN_ * D_) return;
  const int row = idx >> 7, d = idx & 127;
  float h3  = h2[idx] + moa[(long)row * 2 * D_ + d] + mob[(long)row * 2 * D_ + d];
  float m2a = moa[(long)row * 2 * D_ + D_ + d];
  float m2b = mob[(long)row * 2 * D_ + D_ + d];
  float* o = out + (long)idx * 9;
  o[0] = h3;
#pragma unroll
  for (int m = 0; m < 3; ++m) o[1 + m] = x1a[(long)idx * 3 + m] + p1[(long)idx * 3 + m] * m2a;
#pragma unroll
  for (int m = 0; m < 5; ++m) o[4 + m] = x2a[(long)idx * 5 + m] + p2[(long)idx * 5 + m] * m2b;
}

// ---------------------------------------------------------------------------
static void gemm(hipStream_t st, const float* A, const float* Bm, const float* bias,
                 const float* Cin, float* C, int M, int N, int K,
                 int aRS, int aCS, int bRS, int bCS, int cRS, int cCS,
                 int batches, int zdiv,
                 long aB0, long aB1, long bB0, long bB1, long cB0, long cB1, int flags)
{
  dim3 g((M + 15) / 16, (N + 15) / 16, batches), b(32);
  hipLaunchKernelGGL(wmma_gemm_f32, g, b, 0, st, A, Bm, bias, Cin, C, M, N, K,
                     aRS, aCS, bRS, bCS, cRS, cCS, aB0, aB1, bB0, bB1, cB0, cB1, zdiv, flags);
}

extern "C" void kernel_launch(void* const* d_in, const int* in_sizes, int n_in,
                              void* d_out, int out_size, void* d_ws, size_t ws_size,
                              hipStream_t stream)
{
  const float* h    = (const float*)d_in[0];
  const float* x1   = (const float*)d_in[1];
  const float* x2   = (const float*)d_in[2];
  const float* t_ij = (const float*)d_in[3];
  const float* r1   = (const float*)d_in[4];
  const float* r2   = (const float*)d_in[5];
  const float* g_i  = (const float*)d_in[6];
  const float* g_j  = (const float*)d_in[7];
  const float* Wq   = (const float*)d_in[8];
  const float* Wk   = (const float*)d_in[9];
  const float* Wv1  = (const float*)d_in[10];
  const float* bv1  = (const float*)d_in[11];
  const float* Wv2  = (const float*)d_in[12];
  const float* bv2  = (const float*)d_in[13];
  const float* Wp1  = (const float*)d_in[14];
  const float* bp1  = (const float*)d_in[15];
  const float* Wp2  = (const float*)d_in[16];
  const float* bp2  = (const float*)d_in[17];
  const float* Wek  = (const float*)d_in[18];
  const float* Wev  = (const float*)d_in[19];
  const float* Wcomb= (const float*)d_in[20];
  const float* Wqh  = (const float*)d_in[21];
  const float* Wk1h = (const float*)d_in[22];
  const float* Wk2h = (const float*)d_in[23];
  const float* Wep  = (const float*)d_in[24];
  const float* Wres = (const float*)d_in[25];
  const float* P1   = (const float*)d_in[26];
  const float* P2   = (const float*)d_in[27];
  const float* Wf1a = (const float*)d_in[28];
  const float* bf1a = (const float*)d_in[29];
  const float* Wf2a = (const float*)d_in[30];
  const float* bf2a = (const float*)d_in[31];
  const float* Wf1b = (const float*)d_in[32];
  const float* bf1b = (const float*)d_in[33];
  const float* Wf2b = (const float*)d_in[34];
  const float* bf2b = (const float*)d_in[35];
  float* out = (float*)d_out;
  float* ws  = (float*)d_ws;

  size_t o = 0;
  float* HI  = ws + o; o += BN_ * D_;
  float* HJ  = ws + o; o += BN_ * D_;
  float* Q   = ws + o; o += BN_ * DI_;
  float* KK  = ws + o; o += BN_ * DI_;
  float* TV  = ws + o; o += BN_ * DMI_;
  float* VV  = ws + o; o += BN_ * SDI_;
  float* TP  = ws + o; o += BN_ * DMI_;
  float* PVV = ws + o; o += BN_ * SDI_;
  float* Q1  = ws + o; o += BN_ * DE_ * 3;
  float* K1  = ws + o; o += BN_ * DE_ * 3;
  float* Q2  = ws + o; o += BN_ * DE_ * 5;
  float* K2  = ws + o; o += BN_ * DE_ * 5;
  float* WIJ = ws + o; o += (size_t)BNN_ * D_;
  float* TT  = ws + o; o += (size_t)BNN_ * D_;
  float* SN  = ws + o; o += (size_t)B_ * H_ * N_ * N_;
  float* SD  = ws + o; o += (size_t)B_ * H_ * N_ * S_;
  float* H2  = ws + o; o += BN_ * D_;
  float* X1A = ws + o; o += BN_ * D_ * 3;
  float* X2A = ws + o; o += BN_ * D_ * 5;
  float* PJ1 = ws + o; o += BN_ * D_ * 3;
  float* PJ2 = ws + o; o += BN_ * D_ * 5;
  float* IN1 = ws + o; o += BN_ * D_;
  float* IN2 = ws + o; o += BN_ * D_;
  float* FFA = ws + o; o += BN_ * FFD_;
  float* MOA = ws + o; o += BN_ * 2 * D_;
  float* FFB = ws + o; o += BN_ * FFD_;
  float* MOB = ws + o; o += BN_ * 2 * D_;
  bf16* TTB  = (bf16*)(ws + o); o += (size_t)BNN_ * D_ / 2;      // t in bf16
  bf16* WEKT = (bf16*)(ws + o); o += (size_t)D_ * SDI_ / 2;      // Wek^T bf16
  bf16* WEVT = (bf16*)(ws + o); o += (size_t)D_ * SDI_ / 2;      // Wev^T bf16
  bf16* WCOT = (bf16*)(ws + o); o += (size_t)DI_ * D_ / 2;       // Wcomb^T bf16
  (void)ws_size; (void)in_sizes; (void)n_in; (void)out_size;

  // 0) weight conversions (independent of everything else)
  hipLaunchKernelGGL(convT_kernel, dim3((D_*SDI_ + 255)/256), dim3(256), 0, stream, Wek,   WEKT, D_,  SDI_);
  hipLaunchKernelGGL(convT_kernel, dim3((D_*SDI_ + 255)/256), dim3(256), 0, stream, Wev,   WEVT, D_,  SDI_);
  hipLaunchKernelGGL(convT_kernel, dim3((DI_*D_  + 255)/256), dim3(256), 0, stream, Wcomb, WCOT, DI_, D_);

  // 1) LayerNorm
  hipLaunchKernelGGL(ln_kernel, dim3(BN_), dim3(D_), 0, stream, h, g_i, g_j, HI, HJ);

  // 2) q/k head projections (f32 WMMA)
  gemm(stream, HI, Wq, 0, 0, Q,  BN_, DI_, D_, D_,1, DI_,1, DI_,1, 1,1, 0,0,0,0,0,0, 0);
  gemm(stream, HJ, Wk, 0, 0, KK, BN_, DI_, D_, D_,1, DI_,1, DI_,1, 1,1, 0,0,0,0,0,0, 0);

  // 3) v / pv MLPs (f32 WMMA)
  gemm(stream, HJ, Wv1, bv1, 0, TV,  BN_, DMI_, D_,  D_,1,  DMI_,1, DMI_,1, 1,1, 0,0,0,0,0,0, 1);
  gemm(stream, TV, Wv2, bv2, 0, VV,  BN_, SDI_, DMI_, DMI_,1, SDI_,1, SDI_,1, 1,1, 0,0,0,0,0,0, 0);
  gemm(stream, HJ, Wp1, bp1, 0, TP,  BN_, DMI_, D_,  D_,1,  DMI_,1, DMI_,1, 1,1, 0,0,0,0,0,0, 1);
  gemm(stream, TP, Wp2, bp2, 0, PVV, BN_, SDI_, DMI_, DMI_,1, SDI_,1, SDI_,1, 1,1, 0,0,0,0,0,0, 0);

  // 4) q1/k1 (m batched over 3), q2/k2 (batched over 5)
  gemm(stream, x1, Wqh,  0, 0, Q1, BN_, DE_, D_, 3*D_,3, DE_,1, 3*DE_,3, 3,3, 0,1, 0,0, 0,1, 0);
  gemm(stream, x1, Wk1h, 0, 0, K1, BN_, DE_, D_, 3*D_,3, DE_,1, 3*DE_,3, 3,3, 0,1, 0,0, 0,1, 0);
  gemm(stream, x2, Wqh,  0, 0, Q2, BN_, DE_, D_, 5*D_,5, DE_,1, 5*DE_,5, 5,5, 0,1, 0,0, 0,1, 0);
  gemm(stream, x2, Wk2h, 0, 0, K2, BN_, DE_, D_, 5*D_,5, DE_,1, 5*DE_,5, 5,5, 0,1, 0,0, 0,1, 0);

  // 5) w_ij
  hipLaunchKernelGGL(wij_kernel, dim3((BNN_ * D_) / 256), dim3(256), 0, stream, Q1, K1, Q2, K2, WIJ);

  // 6) t = w_ij @ Wep + t_ij @ Wres  (f32 WMMA), then t -> bf16
  gemm(stream, WIJ,  Wep,  0, 0,  TT, BNN_, D_, 2*DE_, 2*DE_,1, D_,1, D_,1, 1,1, 0,0,0,0,0,0, 0);
  gemm(stream, t_ij, Wres, 0, TT, TT, BNN_, D_, D_,    D_,1,    D_,1, D_,1, 1,1, 0,0,0,0,0,0, 2);
  hipLaunchKernelGGL(cvt_bf16_kernel, dim3(((size_t)BNN_*D_)/256), dim3(256), 0, stream,
                     TT, TTB, (long)BNN_*D_);

  // 7) sim_num = q @ k^T, batched over (b,h)  (f32 WMMA)
  gemm(stream, Q, KK, 0, 0, SN, N_, N_, DH_,
       DI_,1, 1,DI_, N_,1,
       B_*H_, H_,
       (long)N_*DI_, DH_, (long)N_*DI_, DH_, (long)H_*N_*N_, (long)N_*N_, 0);

  // 8) init accumulators
  hipLaunchKernelGGL(init_kernel, dim3((BN_ * D_ * 5 + 255) / 256), dim3(256), 0, stream,
                     h, H2, X1A, X2A, SD);

  // 9) fused ek -> sim_den  (bf16 WMMA hot path)
  hipLaunchKernelGGL(simden_kernel, dim3(N_/16, SDI_/16, BN_), dim3(32), 0, stream, TTB, WEKT, Q, SD);

  // 10) fused ev/sea/out + j-reductions  (bf16 WMMA hot path)
  hipLaunchKernelGGL(pass2_kernel, dim3(N_/16, BN_), dim3(256), 0, stream,
                     TTB, WEVT, WCOT, SN, SD, VV, PVV, x1, x2, r1, r2, H2, X1A, X2A);

  // 11) ff branches (f32 WMMA)
  gemm(stream, X1A, P1, 0, 0, PJ1, BN_, D_, D_, 3*D_,3, D_,1, 3*D_,3, 3,3, 0,1, 0,0, 0,1, 0);
  gemm(stream, X2A, P2, 0, 0, PJ2, BN_, D_, D_, 5*D_,5, D_,1, 5*D_,5, 5,5, 0,1, 0,0, 0,1, 0);
  hipLaunchKernelGGL(invnorm_kernel, dim3((BN_*D_+255)/256), dim3(256), 0, stream, PJ1, IN1, 3);
  hipLaunchKernelGGL(invnorm_kernel, dim3((BN_*D_+255)/256), dim3(256), 0, stream, PJ2, IN2, 5);

  gemm(stream, H2,  Wf1a,          0,    0,   FFA, BN_, FFD_, D_, D_,1, FFD_,1, FFD_,1, 1,1, 0,0,0,0,0,0, 0);
  gemm(stream, IN1, Wf1a + (size_t)D_*FFD_, bf1a, FFA, FFA, BN_, FFD_, D_, D_,1, FFD_,1, FFD_,1, 1,1, 0,0,0,0,0,0, 3);
  gemm(stream, FFA, Wf2a, bf2a, 0, MOA, BN_, 2*D_, FFD_, FFD_,1, 2*D_,1, 2*D_,1, 1,1, 0,0,0,0,0,0, 0);

  gemm(stream, H2,  Wf1b,          0,    0,   FFB, BN_, FFD_, D_, D_,1, FFD_,1, FFD_,1, 1,1, 0,0,0,0,0,0, 0);
  gemm(stream, IN2, Wf1b + (size_t)D_*FFD_, bf1b, FFB, FFB, BN_, FFD_, D_, D_,1, FFD_,1, FFD_,1, 1,1, 0,0,0,0,0,0, 3);
  gemm(stream, FFB, Wf2b, bf2b, 0, MOB, BN_, 2*D_, FFD_, FFD_,1, 2*D_,1, 2*D_,1, 1,1, 0,0,0,0,0,0, 0);

  // 12) final assembly
  hipLaunchKernelGGL(final_kernel, dim3((BN_*D_+255)/256), dim3(256), 0, stream,
                     H2, MOA, MOB, X1A, X2A, PJ1, PJ2, out);
}
